// SimGNN_12249246728482
// MI455X (gfx1250) — matrix operand order repaired and checked by reading, not compiled, not run on hardware
//
#include <hip/hip_runtime.h>
#include <hip/hip_bf16.h>

typedef __attribute__((ext_vector_type(16))) _Float16 v16h;
typedef __attribute__((ext_vector_type(8)))  _Float16 v8h;
typedef __attribute__((ext_vector_type(8)))  float    v8f;
typedef __attribute__((ext_vector_type(2)))  float    v2f;

#define BINS 16
#define WAVES 8   // waves per block in hist kernels
#define CT 8      // column tiles per wave in hist kernels

// ---------- small utility kernels ----------

__global__ void fill_f32(float* p, float v, int n) {
    int i = blockIdx.x * blockDim.x + threadIdx.x;
    if (i < n) p[i] = v;
}

__global__ void deg_count(const int* ei, float* deg, int E) {
    int e = blockIdx.x * blockDim.x + threadIdx.x;
    if (e < E) atomicAdd(&deg[ei[E + e]], 1.0f);   // tgt = ei[1][e]
}

__global__ void deg_to_dinv(float* deg, int N) {
    int i = blockIdx.x * blockDim.x + threadIdx.x;
    if (i < N) deg[i] = rsqrtf(deg[i] + 1.0f);     // +1 self loop; deg>=1 always
}

// ---------- dense transform: Y[N,NT*16] = X[N,Kin] @ W[Kin,NT*16] via f32 WMMA ----------
// One wave owns a 16-row strip of X and all NT column tiles (accumulators in VGPRs),
// looping K in steps of 4 with V_WMMA_F32_16X16X4_F32.  X is read exactly once.
// NT is compile-time: no branches between WMMAs.  Requires N%16==0, Kin%4==0.
template <int NT>
__global__ void gemm_xw_wmma(const float* __restrict__ X, const float* __restrict__ W,
                             float* __restrict__ Y, int N, int Kin) {
#if defined(__gfx1250__) && __has_builtin(__builtin_amdgcn_wmma_f32_16x16x4_f32)
    const int Kout = NT * 16;
    int lane = threadIdx.x & 31;
    int wave = threadIdx.x >> 5;
    int m0 = (blockIdx.x * (blockDim.x >> 5) + wave) * 16;
    if (m0 >= N) return;                       // wave-uniform: EXEC stays all-1s for WMMA
    int row  = m0 + (lane & 15);
    int koff = (lane >> 4) * 2;                // 32-bit A/B layout: lanes 16-31 hold K+2,K+3
    int col  = lane & 15;
    v8f acc[NT];
    #pragma unroll
    for (int nt = 0; nt < NT; ++nt) acc[nt] = (v8f){0.f,0.f,0.f,0.f,0.f,0.f,0.f,0.f};
    const float* xr = X + (size_t)row * Kin;
    for (int k0 = 0; k0 < Kin; k0 += 4) {
        v2f a = *(const v2f*)(xr + k0 + koff); // adjacent pair -> single b64 load
        const float* w0 = W + (size_t)(k0 + koff) * Kout + col;
        const float* w1 = w0 + Kout;
        #pragma unroll
        for (int nt = 0; nt < NT; ++nt) {
            v2f b;
            b[0] = w0[nt * 16];
            b[1] = w1[nt * 16];
            acc[nt] = __builtin_amdgcn_wmma_f32_16x16x4_f32(false, a, false, b,
                                                            (short)0, acc[nt], false, false);
        }
    }
    int mrow = (lane >> 4) * 8;                // D layout: VGPR r, lane<16 -> M=r, lane>=16 -> M=8+r
    #pragma unroll
    for (int nt = 0; nt < NT; ++nt)
        #pragma unroll
        for (int r = 0; r < 8; ++r)
            Y[(size_t)(m0 + mrow + r) * Kout + nt * 16 + col] = acc[nt][r];
#else
    // fallback (also what the host pass parses): plain scalar GEMM
    const int Kout = NT * 16;
    int total = N * Kout;
    for (int idx = blockIdx.x * blockDim.x + threadIdx.x; idx < total;
         idx += gridDim.x * blockDim.x) {
        int r = idx / Kout, c = idx - r * Kout;
        const float* xr = X + (size_t)r * Kin;
        float accs = 0.f;
        for (int k = 0; k < Kin; ++k) accs += xr[k] * W[(size_t)k * Kout + c];
        Y[idx] = accs;
    }
#endif
}

// generic scalar GEMM for unexpected shapes
__global__ void gemm_xw_scalar(const float* __restrict__ X, const float* __restrict__ W,
                               float* __restrict__ Y, int N, int Kin, int Kout) {
    int total = N * Kout;
    for (int idx = blockIdx.x * blockDim.x + threadIdx.x; idx < total;
         idx += gridDim.x * blockDim.x) {
        int r = idx / Kout, c = idx - r * Kout;
        const float* xr = X + (size_t)r * Kin;
        float accs = 0.f;
        for (int k = 0; k < Kin; ++k) accs += xr[k] * W[(size_t)k * Kout + c];
        Y[idx] = accs;
    }
}

// ---------- edge scatter: agg[tgt] += dinv[s]*dinv[t] * tmp[src] ----------
__global__ void scatter_edges(const int* __restrict__ ei, const float* __restrict__ dinv,
                              const float* __restrict__ tmp, float* __restrict__ agg,
                              int E, int F) {
    int idx = blockIdx.x * blockDim.x + threadIdx.x;
    int tot = E * F;
    if (idx >= tot) return;
    int e = idx / F, f = idx - e * F;
    int s = ei[e], t = ei[E + e];
    float nrm = dinv[s] * dinv[t];
    atomicAdd(&agg[(size_t)t * F + f], nrm * tmp[(size_t)s * F + f]);
}

// self-loop contribution + bias + optional relu
__global__ void finalize(const float* __restrict__ agg, const float* __restrict__ tmp,
                         const float* __restrict__ dinv, const float* __restrict__ b,
                         float* __restrict__ out, int N, int F, int do_relu) {
    int idx = blockIdx.x * blockDim.x + threadIdx.x;
    if (idx >= N * F) return;
    int i = idx / F, f = idx - i * F;
    float d = dinv[i];
    float v = agg[idx] + d * d * tmp[idx] + b[f];
    out[idx] = do_relu ? fmaxf(v, 0.f) : v;
}

__global__ void to_half(const float* __restrict__ x, _Float16* __restrict__ y, int n) {
    int i = blockIdx.x * blockDim.x + threadIdx.x;
    if (i < n) y[i] = (_Float16)x[i];
}

// ---------- attention pooling ----------
__global__ void colmean32(const float* __restrict__ af, float* __restrict__ cm, int N) {
    int c = blockIdx.x;                // 0..31
    float s = 0.f;
    for (int i = threadIdx.x; i < N; i += blockDim.x) s += af[(size_t)i * 32 + c];
    __shared__ float red[256];
    red[threadIdx.x] = s;
    __syncthreads();
    for (int st = 128; st > 0; st >>= 1) {
        if (threadIdx.x < st) red[threadIdx.x] += red[threadIdx.x + st];
        __syncthreads();
    }
    if (threadIdx.x == 0) cm[c] = red[0] / (float)N;
}

__global__ void ctx32(const float* __restrict__ cm, const float* __restrict__ Wa,
                      float* __restrict__ ctx) {
    int j = threadIdx.x;               // 32 threads
    float a = 0.f;
    for (int k = 0; k < 32; ++k) a += cm[k] * Wa[k * 32 + j];
    ctx[j] = tanhf(a);
}

__global__ void attpool(const float* __restrict__ af, const float* __restrict__ ctx,
                        float* __restrict__ pooled, int N) {
    __shared__ float part[32];
    if (threadIdx.x < 32) part[threadIdx.x] = 0.f;
    __syncthreads();
    int i = blockIdx.x * blockDim.x + threadIdx.x;
    if (i < N) {
        float dot = 0.f;
        #pragma unroll
        for (int j = 0; j < 32; ++j) dot += af[(size_t)i * 32 + j] * ctx[j];
        float att = 1.f / (1.f + expf(-dot));
        #pragma unroll
        for (int j = 0; j < 32; ++j) atomicAdd(&part[j], af[(size_t)i * 32 + j] * att);
    }
    __syncthreads();
    if (threadIdx.x < 32) atomicAdd(&pooled[threadIdx.x], part[threadIdx.x]);
}

// ---------- order-preserving float<->uint for atomic min/max ----------
__device__ __forceinline__ unsigned fenc(float f) {
    unsigned u = __float_as_uint(f);
    return (u & 0x80000000u) ? ~u : (u | 0x80000000u);
}
__device__ __forceinline__ float fdec(unsigned e) {
    unsigned u = (e & 0x80000000u) ? (e & 0x7FFFFFFFu) : ~e;
    return __uint_as_float(u);
}

// ---------- WMMA fragment loads (ISA 7.12.2 layouts) ----------
// A 16x32 f16: lane l<16 -> row m0+l, K{0..7,16..23}; lane l>=16 -> row m0+l-16, K{8..15,24..31}
__device__ __forceinline__ v16h load_a_frag(const _Float16* __restrict__ af, int m0, int lane) {
    int row = m0 + (lane & 15);
    int k0  = (lane >> 4) * 8;
    const _Float16* p = af + (size_t)row * 32 + k0;
    v8h lo = *(const v8h*)(p);
    v8h hi = *(const v8h*)(p + 16);
    v16h r;
    #pragma unroll
    for (int i = 0; i < 8; ++i) { r[i] = lo[i]; r[i + 8] = hi[i]; }
    return r;
}
// B 32x16 f16 (B[k][n] = af2[n][k]): lane l<16 -> col n0+l, K{0..7,16..23}; lane l>=16 -> col n0+l-16, K{8..15,24..31}
__device__ __forceinline__ v16h load_b_frag(const _Float16* __restrict__ bf, int n0, int lane) {
    int col = n0 + (lane & 15);
    int k0  = (lane >> 4) * 8;
    const _Float16* p = bf + (size_t)col * 32 + k0;
    v8h lo = *(const v8h*)(p);
    v8h hi = *(const v8h*)(p + 16);
    v16h r;
    #pragma unroll
    for (int i = 0; i < 8; ++i) { r[i] = lo[i]; r[i + 8] = hi[i]; }
    return r;
}

__device__ __forceinline__ v8f tile_wmma(const v16h& a, const _Float16* __restrict__ B,
                                         int n0, int lane) {
    v16h b = load_b_frag(B, n0, lane);
    v8f c = {};
    c = __builtin_amdgcn_wmma_f32_16x16x32_f16(false, a, false, b, (short)0, c,
                                               false, false);
    return c;
}

// ---------- pass 1: min/max of af1 @ af2^T ----------
__global__ void hist_minmax_wmma(const _Float16* __restrict__ A, const _Float16* __restrict__ B,
                                 unsigned* __restrict__ mnmx) {
    int lane = threadIdx.x & 31, wave = threadIdx.x >> 5;
    int m0 = blockIdx.x * 16;
    int n0 = (blockIdx.y * WAVES + wave) * 16 * CT;
    v16h a = load_a_frag(A, m0, lane);
    float lmn = 3.4e38f, lmx = -3.4e38f;
    #pragma unroll
    for (int t = 0; t < CT; ++t) {
        v8f c = tile_wmma(a, B, n0 + t * 16, lane);
        #pragma unroll
        for (int r = 0; r < 8; ++r) {
            float v = c[r];
            lmn = fminf(lmn, v);
            lmx = fmaxf(lmx, v);
        }
    }
    __shared__ unsigned smn, smx;
    if (threadIdx.x == 0) { smn = 0xFFFFFFFFu; smx = 0u; }
    __syncthreads();
    atomicMin(&smn, fenc(lmn));
    atomicMax(&smx, fenc(lmx));
    __syncthreads();
    if (threadIdx.x == 0) {
        atomicMin(&mnmx[0], smn);
        atomicMax(&mnmx[1], smx);
    }
}

__global__ void init_mnmx(unsigned* m) { m[0] = 0xFFFFFFFFu; m[1] = 0u; }

__global__ void hist_prep(const unsigned* __restrict__ mnmx, float* __restrict__ mnsc,
                          unsigned* __restrict__ hist) {
    float mn = fdec(mnmx[0]), mx = fdec(mnmx[1]);
    float w = (mx > mn) ? (mx - mn) : 1.f;
    mnsc[0] = mn;
    mnsc[1] = (float)BINS / w;
    for (int i = 0; i < BINS; ++i) hist[i] = 0u;
}

// ---------- pass 2: recompute tiles, bin into histogram ----------
__global__ void hist_bin_wmma(const _Float16* __restrict__ A, const _Float16* __restrict__ B,
                              const float* __restrict__ mnsc, unsigned* __restrict__ hist) {
    __shared__ unsigned sh[BINS];
    if (threadIdx.x < BINS) sh[threadIdx.x] = 0u;
    __syncthreads();
    int lane = threadIdx.x & 31, wave = threadIdx.x >> 5;
    int m0 = blockIdx.x * 16;
    int n0 = (blockIdx.y * WAVES + wave) * 16 * CT;
    float mn = mnsc[0], sc = mnsc[1];
    v16h a = load_a_frag(A, m0, lane);
    #pragma unroll
    for (int t = 0; t < CT; ++t) {
        v8f c = tile_wmma(a, B, n0 + t * 16, lane);
        #pragma unroll
        for (int r = 0; r < 8; ++r) {
            int ix = (int)((c[r] - mn) * sc);
            ix = min(BINS - 1, max(0, ix));
            atomicAdd(&sh[ix], 1u);
        }
    }
    __syncthreads();
    if (threadIdx.x < BINS) atomicAdd(&hist[threadIdx.x], sh[threadIdx.x]);
}

// ---------- NTN + histogram normalize + MLP head (trivial size) ----------
__global__ void head(const float* __restrict__ p1, const float* __restrict__ p2,
                     const float* __restrict__ Wt, const float* __restrict__ V,
                     const float* __restrict__ bt, const unsigned* __restrict__ hist,
                     const float* __restrict__ fc1W, const float* __restrict__ fc1b,
                     const float* __restrict__ fc2W, const float* __restrict__ fc2b,
                     const float* __restrict__ fc3W, const float* __restrict__ fc3b,
                     const float* __restrict__ scW, const float* __restrict__ scb,
                     const float* __restrict__ avgv, float* __restrict__ out) {
    float feat[32];
    for (int k = 0; k < 16; ++k) {
        float s = 0.f;
        for (int d = 0; d < 32; ++d) {
            float f1 = p1[d];
            for (int e = 0; e < 32; ++e) s += f1 * Wt[(d * 32 + e) * 16 + k] * p2[e];
        }
        float bl = 0.f;
        for (int j = 0; j < 32; ++j) bl += V[k * 64 + j] * p1[j];
        for (int j = 0; j < 32; ++j) bl += V[k * 64 + 32 + j] * p2[j];
        feat[k] = fmaxf(s + bl + bt[k], 0.f);
    }
    float tot = 0.f;
    for (int i = 0; i < BINS; ++i) tot += (float)hist[i];
    for (int i = 0; i < BINS; ++i) feat[16 + i] = (float)hist[i] / tot;
    float h1[16], h2[8], h3[4];
    for (int j = 0; j < 16; ++j) {
        float a = fc1b[j];
        for (int i = 0; i < 32; ++i) a += feat[i] * fc1W[i * 16 + j];
        h1[j] = fmaxf(a, 0.f);
    }
    for (int j = 0; j < 8; ++j) {
        float a = fc2b[j];
        for (int i = 0; i < 16; ++i) a += h1[i] * fc2W[i * 8 + j];
        h2[j] = fmaxf(a, 0.f);
    }
    for (int j = 0; j < 4; ++j) {
        float a = fc3b[j];
        for (int i = 0; i < 8; ++i) a += h2[i] * fc3W[i * 4 + j];
        h3[j] = fmaxf(a, 0.f);
    }
    float a = scb[0];
    for (int i = 0; i < 4; ++i) a += h3[i] * scW[i];
    float score = 1.f / (1.f + expf(-a));
    out[0] = score;
    out[1] = -logf(score) * avgv[0];
}

// ---------- host orchestration ----------
static inline int cdiv(int a, int b) { return (a + b - 1) / b; }

static void launch_gemm(const float* X, const float* W, float* Y, int N, int Kin, int Kout,
                        hipStream_t stream) {
    int blocks = cdiv(N / 16, 4);   // 4 waves / block, one 16-row strip per wave
    switch (Kout >> 4) {
        case 8: gemm_xw_wmma<8><<<blocks, 128, 0, stream>>>(X, W, Y, N, Kin); break;
        case 4: gemm_xw_wmma<4><<<blocks, 128, 0, stream>>>(X, W, Y, N, Kin); break;
        case 2: gemm_xw_wmma<2><<<blocks, 128, 0, stream>>>(X, W, Y, N, Kin); break;
        case 1: gemm_xw_wmma<1><<<blocks, 128, 0, stream>>>(X, W, Y, N, Kin); break;
        default:
            gemm_xw_scalar<<<4096, 256, 0, stream>>>(X, W, Y, N, Kin, Kout);
            break;
    }
}

extern "C" void kernel_launch(void* const* d_in, const int* in_sizes, int n_in,
                              void* d_out, int out_size, void* d_ws, size_t ws_size,
                              hipStream_t stream) {
    const int*   ei1  = (const int*)d_in[0];
    const int*   ei2  = (const int*)d_in[1];
    const float* x1   = (const float*)d_in[2];
    const float* x2   = (const float*)d_in[3];
    const float* avgv = (const float*)d_in[4];
    const float* W1 = (const float*)d_in[5],  *b1 = (const float*)d_in[6];
    const float* W2 = (const float*)d_in[7],  *b2 = (const float*)d_in[8];
    const float* W3 = (const float*)d_in[9],  *b3 = (const float*)d_in[10];
    const float* Wa = (const float*)d_in[11];
    const float* Wt = (const float*)d_in[12], *V  = (const float*)d_in[13];
    const float* bt = (const float*)d_in[14];
    const float* fc1W = (const float*)d_in[15], *fc1b = (const float*)d_in[16];
    const float* fc2W = (const float*)d_in[17], *fc2b = (const float*)d_in[18];
    const float* fc3W = (const float*)d_in[19], *fc3b = (const float*)d_in[20];
    const float* scW  = (const float*)d_in[21], *scb  = (const float*)d_in[22];

    const int E  = in_sizes[0] / 2;
    const int F1 = in_sizes[6];           // 128
    const int F2 = in_sizes[8];           // 64
    const int F3 = in_sizes[10];          // 32
    const int L  = in_sizes[5] / F1;      // 64
    const int N  = in_sizes[2] / L;       // 8192

    // workspace carve-up (floats)
    float* ws    = (float*)d_ws;
    float* dinv1 = ws;
    float* dinv2 = dinv1 + N;
    float* af1   = dinv2 + N;
    float* af2   = af1 + (size_t)N * F3;
    _Float16* af1h = (_Float16*)(af2 + (size_t)N * F3);
    _Float16* af2h = af1h + (size_t)N * F3;
    float* smallb = (float*)(af2h + (size_t)N * F3);
    float* cm   = smallb;          // [32]
    float* ctx  = smallb + 32;     // [32]
    float* p1   = smallb + 64;     // [32]
    float* p2   = smallb + 96;     // [32]
    unsigned* mnmx = (unsigned*)(smallb + 128);  // [2]
    unsigned* hist = mnmx + 2;                   // [16]
    float* mnsc = (float*)(hist + 16);           // [2]
    float* tmp  = smallb + 256;                  // [N,128]
    float* agg  = tmp + (size_t)N * F1;          // [N,128]
    float* hb   = agg + (size_t)N * F1;          // [N,128]

    const int TB = 256;

    auto run_graph = [&](const int* ei, const float* x, float* dinv, float* af,
                         _Float16* afh, float* pooled) {
        // degree -> dinv
        fill_f32<<<cdiv(N, TB), TB, 0, stream>>>(dinv, 0.f, N);
        deg_count<<<cdiv(E, TB), TB, 0, stream>>>(ei, dinv, E);
        deg_to_dinv<<<cdiv(N, TB), TB, 0, stream>>>(dinv, N);

        const float* in = x;  int Kin = L;
        const float* Ws[3] = {W1, W2, W3};
        const float* bs[3] = {b1, b2, b3};
        const int    Ko[3] = {F1, F2, F3};
        for (int layer = 0; layer < 3; ++layer) {
            int Kout = Ko[layer];
            int total = N * Kout;
            launch_gemm(in, Ws[layer], tmp, N, Kin, Kout, stream);
            fill_f32<<<cdiv(total, TB), TB, 0, stream>>>(agg, 0.f, total);
            scatter_edges<<<cdiv(E * Kout, TB), TB, 0, stream>>>(ei, dinv, tmp, agg, E, Kout);
            float* dst = (layer == 2) ? af : hb;
            finalize<<<cdiv(total, TB), TB, 0, stream>>>(agg, tmp, dinv, bs[layer], dst, N, Kout,
                                                         layer == 2 ? 0 : 1);
            in = dst; Kin = Kout;
        }
        to_half<<<cdiv(N * F3, TB), TB, 0, stream>>>(af, afh, N * F3);

        // attention pooling
        colmean32<<<32, 256, 0, stream>>>(af, cm, N);
        ctx32<<<1, 32, 0, stream>>>(cm, Wa, ctx);
        fill_f32<<<1, 32, 0, stream>>>(pooled, 0.f, 32);
        attpool<<<cdiv(N, TB), TB, 0, stream>>>(af, ctx, pooled, N);
    };

    run_graph(ei1, x1, dinv1, af1, af1h, p1);
    run_graph(ei2, x2, dinv2, af2, af2h, p2);

    // histogram of af1 @ af2^T via f16 WMMA, two passes (minmax, then bin)
    dim3 hgrid(N / 16, N / (16 * CT * WAVES));
    init_mnmx<<<1, 1, 0, stream>>>(mnmx);
    hist_minmax_wmma<<<hgrid, 32 * WAVES, 0, stream>>>(af1h, af2h, mnmx);
    hist_prep<<<1, 1, 0, stream>>>(mnmx, mnsc, hist);
    hist_bin_wmma<<<hgrid, 32 * WAVES, 0, stream>>>(af1h, af2h, mnsc, hist);

    head<<<1, 1, 0, stream>>>(p1, p2, Wt, V, bt, hist, fc1W, fc1b, fc2W, fc2b,
                              fc3W, fc3b, scW, scb, avgv, (float*)d_out);
}
